// PredictiveReasoningBlock_42236708389241
// MI455X (gfx1250) — matrix operand order repaired and checked by reading, not compiled, not run on hardware
//
#include <hip/hip_runtime.h>
#include <math.h>

// ---------------------------------------------------------------------------
// PredictiveReasoningBlock forward for MI455X (gfx1250, wave32, WMMA).
// Channels-last activations (B,T,L,C); all channel GEMMs and 3x3 convs run
// through v_wmma_f32_16x16x32_f16 (f16 inputs, fp32 accumulate).
// ---------------------------------------------------------------------------

#define BB 512
#define TT 4
#define LLEN 24
#define CC 32
#define NHEAD 8

typedef __attribute__((ext_vector_type(16))) _Float16 v16h;
typedef __attribute__((ext_vector_type(8)))  float    v8f;

#define EP_BIAS  1   // + bias[col]
#define EP_SCALE 2   // * gain[col] + bias[col]   (cna epilogue)
#define EP_RELU  4
#define EP_GELU  8
#define EP_ADD  16   // + add[row*ldadd + col]    (residual, applied last)

__device__ __forceinline__ float geluf(float x) {
    return 0.5f * x * (1.0f + erff(x * 0.70710678118654752f));
}

// Convert two contiguous 8-float chunks (at p and p+8 floats) into the
// 16-element f16 A-operand fragment for this lane.
__device__ __forceinline__ v16h load_a_frag(const float* __restrict__ p) {
    const float4* q0 = (const float4*)p;
    const float4* q1 = (const float4*)(p + 8);
    float4 c0 = q0[0], c1 = q0[1], c2 = q1[0], c3 = q1[1];
    v16h a;
    a[0]  = (_Float16)c0.x; a[1]  = (_Float16)c0.y;
    a[2]  = (_Float16)c0.z; a[3]  = (_Float16)c0.w;
    a[4]  = (_Float16)c1.x; a[5]  = (_Float16)c1.y;
    a[6]  = (_Float16)c1.z; a[7]  = (_Float16)c1.w;
    a[8]  = (_Float16)c2.x; a[9]  = (_Float16)c2.y;
    a[10] = (_Float16)c2.z; a[11] = (_Float16)c2.w;
    a[12] = (_Float16)c3.x; a[13] = (_Float16)c3.y;
    a[14] = (_Float16)c3.z; a[15] = (_Float16)c3.w;
    return a;
}

// ---------------------------------------------------------------------------
// Generic WMMA row-GEMM:  Y[n, colOff+o] = epilogue( sum_i X[n,i] * W[o,i] )
// One wave = one 16x16 tile. I in {32,64,128}, O in {32,64} (multiples of 16).
// Weights staged in LDS as f16. Fully branch-free inner loop.
// ---------------------------------------------------------------------------
__global__ void k_gemm(const float* __restrict__ X, int ldx,
                       const float* __restrict__ Wt,
                       const float* __restrict__ bias,
                       const float* __restrict__ gain,
                       const float* __restrict__ add, int ldadd,
                       float* __restrict__ Y, int ldy, int colOff,
                       int Nrow, int I, int O, int flags) {
    __shared__ _Float16 sW[4096];
    const int nW = O * I;
    for (int i = threadIdx.x; i < nW; i += blockDim.x) sW[i] = (_Float16)Wt[i];
    __syncthreads();

    const int wave = (blockIdx.x * blockDim.x + threadIdx.x) >> 5;
    const int lane = threadIdx.x & 31;
    const int rowTiles = Nrow >> 4;
    const int colTiles = O >> 4;
    if (wave >= rowTiles * colTiles) return;           // wave-uniform exit
    const int rt = wave / colTiles, ctl = wave % colTiles;
    const int h = lane >> 4, n16 = lane & 15;
    const int arow = rt * 16 + n16;                    // A-row for this lane
    const int ocol = ctl * 16 + n16;                   // B-col for this lane
    const float* __restrict__ xrow = X + (size_t)arow * ldx;
    const _Float16* __restrict__ brow = sW + ocol * I + 16 * h;
    __builtin_prefetch(xrow, 0, 1);

    v8f acc = {};
    for (int kb = 0; kb < I; kb += 32) {
        v16h a = load_a_frag(xrow + kb + 16 * h);
        v16h b = *(const v16h*)(brow + kb);
        acc = __builtin_amdgcn_wmma_f32_16x16x32_f16(false, a, false, b,
                                                     (short)0, acc, false, false);
    }
#pragma unroll
    for (int r = 0; r < 8; r++) {
        int orow = rt * 16 + r + 8 * h;
        float v = acc[r];
        if (flags & EP_BIAS)  v += bias[ocol];
        if (flags & EP_SCALE) v = v * gain[ocol] + bias[ocol];
        if (flags & EP_GELU)  v = geluf(v);
        if (flags & EP_RELU)  v = fmaxf(v, 0.0f);
        if (flags & EP_ADD)   v += add[(size_t)orow * ldadd + ocol];
        Y[(size_t)orow * ldy + colOff + ocol] = v;
    }
}

// ---------------------------------------------------------------------------
// Conv-weight repack: OIHW f32 -> zero-padded f16 [CoutP x Kp], K-major with
// K = tap*Cin + ci (tap = (ky+1)*3 + (kx+1)), matching the im2col K order.
// ---------------------------------------------------------------------------
__global__ void k_packw(const float* __restrict__ Wt, _Float16* __restrict__ Wp,
                        int Cin, int Cout, int Kp, int total) {
    int tid = blockIdx.x * blockDim.x + threadIdx.x;
    if (tid >= total) return;
    int k = tid % Kp, co = tid / Kp;
    float v = 0.0f;
    if (co < Cout && k < 9 * Cin) {
        int tap = k / Cin, ci = k - tap * Cin;
        v = Wt[((size_t)co * Cin + ci) * 9 + tap];
    }
    Wp[tid] = (_Float16)v;
}

// ---------------------------------------------------------------------------
// WMMA implicit-GEMM 3x3 conv, pad=1, Cin % 32 == 0 (32/64/128 channels).
// Each 32-wide K chunk = one tap x 32 contiguous channels -> vector loads with
// a single per-lane border predicate per tap. B from packed f16 weights.
// ---------------------------------------------------------------------------
__global__ void k_conv3_c32(const float* __restrict__ X, const _Float16* __restrict__ Wp,
                            const float* __restrict__ gain, const float* __restrict__ bias,
                            const float* __restrict__ add, float* __restrict__ Y,
                            int NB, int H, int Wd, int Cin, int Cout, int Kp, int flags) {
    const int rows = NB * H * Wd;
    const int rowTiles = rows >> 4;
    const int colTiles = ((Cout + 15) & ~15) >> 4;
    const int wave = (blockIdx.x * blockDim.x + threadIdx.x) >> 5;
    const int lane = threadIdx.x & 31;
    if (wave >= rowTiles * colTiles) return;
    const int rt = wave / colTiles, ctl = wave % colTiles;
    const int h = lane >> 4, n16 = lane & 15;
    const int ridx = rt * 16 + n16;
    const int x0 = ridx % Wd;
    const int y0 = (ridx / Wd) % H;
    const int nb = ridx / (Wd * H);
    const int co = ctl * 16 + n16;
    const _Float16* __restrict__ brow = Wp + (size_t)co * Kp + 16 * h;

    v8f acc = {};
#pragma unroll
    for (int tap = 0; tap < 9; tap++) {
        const int ky = tap / 3 - 1, kx = tap % 3 - 1;
        const int yy = y0 + ky, xx = x0 + kx;
        const bool ok = (yy >= 0) && (yy < H) && (xx >= 0) && (xx < Wd);
        const float* __restrict__ base = X + ((size_t)(nb * H + yy) * Wd + xx) * Cin;
        for (int coff = 0; coff < Cin; coff += 32) {
            v16h a = {};
            if (ok) a = load_a_frag(base + coff + 16 * h);
            v16h b = *(const v16h*)(brow + tap * Cin + coff);
            acc = __builtin_amdgcn_wmma_f32_16x16x32_f16(false, a, false, b,
                                                         (short)0, acc, false, false);
        }
    }
#pragma unroll
    for (int r = 0; r < 8; r++) {
        int orow = rt * 16 + r + 8 * h;
        if (co >= Cout) continue;
        float v = acc[r];
        v = v * gain[co] + bias[co];
        if (flags & EP_RELU) v = fmaxf(v, 0.0f);
        if (add) v += add[(size_t)orow * Cout + co];
        Y[(size_t)orow * Cout + co] = v;
    }
}

// ---------------------------------------------------------------------------
// WMMA implicit-GEMM 3x3 conv for small power-of-two Cin (4 / 16): gather A
// per element (shift/mask indexing), branch-free packed-f16 B.
// ---------------------------------------------------------------------------
__global__ void k_conv3_small(const float* __restrict__ X, const _Float16* __restrict__ Wp,
                              const float* __restrict__ gain, const float* __restrict__ bias,
                              const float* __restrict__ add, float* __restrict__ Y,
                              int NB, int H, int Wd, int Cin, int Cout, int Kp, int lg,
                              int flags) {
    const int K = 9 * Cin;
    const int rows = NB * H * Wd;
    const int rowTiles = rows >> 4;
    const int colTiles = ((Cout + 15) & ~15) >> 4;
    const int wave = (blockIdx.x * blockDim.x + threadIdx.x) >> 5;
    const int lane = threadIdx.x & 31;
    if (wave >= rowTiles * colTiles) return;
    const int rt = wave / colTiles, ctl = wave % colTiles;
    const int h = lane >> 4, n16 = lane & 15;
    const int ridx = rt * 16 + n16;
    const int x0 = ridx % Wd;
    const int y0 = (ridx / Wd) % H;
    const int nb = ridx / (Wd * H);
    const int co = ctl * 16 + n16;
    const int cmask = Cin - 1;
    const _Float16* __restrict__ brow = Wp + (size_t)co * Kp + 16 * h;

    v8f acc = {};
    for (int kb = 0; kb < Kp; kb += 32) {
        v16h a = {};
#pragma unroll
        for (int e = 0; e < 16; e++) {
            int v = e >> 1, p = e & 1;
            int ka = kb + 16 * h + 2 * v + p;
            float av = 0.0f;
            if (ka < K) {
                int tap = ka >> lg, ci = ka & cmask;
                int ky = tap / 3 - 1, kx = tap % 3 - 1;
                int yy = y0 + ky, xx = x0 + kx;
                if (yy >= 0 && yy < H && xx >= 0 && xx < Wd)
                    av = X[((size_t)(nb * H + yy) * Wd + xx) * Cin + ci];
            }
            a[e] = (_Float16)av;
        }
        v16h b = *(const v16h*)(brow + kb);
        acc = __builtin_amdgcn_wmma_f32_16x16x32_f16(false, a, false, b,
                                                     (short)0, acc, false, false);
    }
#pragma unroll
    for (int r = 0; r < 8; r++) {
        int orow = rt * 16 + r + 8 * h;
        if (co >= Cout) continue;
        float v = acc[r];
        v = v * gain[co] + bias[co];
        if (flags & EP_RELU) v = fmaxf(v, 0.0f);
        if (add) v += add[(size_t)orow * Cout + co];
        Y[(size_t)orow * Cout + co] = v;
    }
}

// ---------------------------------------------------------------------------
// Elementwise / normalization / attention helper kernels (VALU)
// ---------------------------------------------------------------------------
__global__ void k_proj_in(const float* __restrict__ x, const float* __restrict__ mW,
                          const float* __restrict__ mb, float* __restrict__ x0, int total) {
    int tid = blockIdx.x * blockDim.x + threadIdx.x;
    if (tid >= total) return;
    int c = tid & 31;
    int l = (tid >> 5) % LLEN;
    int bt = tid / (32 * LLEN);
    int t = bt & 3, b = bt >> 2;
    const float* xr = x + (((size_t)(b * 32 + c) * 4 + t) * 25);
    float acc = mb[l];
    const float* wr = mW + l * 25;
#pragma unroll
    for (int q = 0; q < 25; q++) acc += xr[q] * wr[q];
    x0[tid] = acc;
}

__global__ void k_copy(const float* __restrict__ a, float* __restrict__ y, int n) {
    int i = blockIdx.x * blockDim.x + threadIdx.x;
    if (i < n) y[i] = a[i];
}

__global__ void k_relu_sub(const float* __restrict__ a, const float* __restrict__ b,
                           float* __restrict__ y, int n) {
    int i = blockIdx.x * blockDim.x + threadIdx.x;
    if (i < n) y[i] = fmaxf(a[i] - b[i], 0.0f);
}

__global__ void k_gelumul(const float* __restrict__ a, const float* __restrict__ b,
                          float* __restrict__ y, int n) {
    int i = blockIdx.x * blockDim.x + threadIdx.x;
    if (i < n) y[i] = geluf(a[i]) * b[i];
}

__global__ void k_l2n_row(const float* __restrict__ x, float* __restrict__ y, int nrow) {
    int row = blockIdx.x * blockDim.x + threadIdx.x;
    if (row >= nrow) return;
    const float* xr = x + (size_t)row * 32;
    float s = 0.0f;
#pragma unroll
    for (int c = 0; c < 32; c++) s += xr[c] * xr[c];
    float inv = 1.0f / fmaxf(sqrtf(s), 1e-12f);
    float* yr = y + (size_t)row * 32;
#pragma unroll
    for (int c = 0; c < 32; c++) yr[c] = xr[c] * inv;
}

// per-head (D=4) l2norm with optional elementwise gate (tc)
__global__ void k_l2n_head(const float* __restrict__ x, int ldx, int offx,
                           const float* __restrict__ mul,
                           float* __restrict__ y, int ldy, int offy, int nrow) {
    int tid = blockIdx.x * blockDim.x + threadIdx.x;
    if (tid >= nrow * NHEAD) return;
    int g = tid & 7, row = tid >> 3;
    const float4 v4 = *(const float4*)(x + (size_t)row * ldx + offx + g * 4);
    float a = v4.x, b = v4.y, c = v4.z, d = v4.w;
    float inv = 1.0f / fmaxf(sqrtf(a * a + b * b + c * c + d * d), 1e-12f);
    float* yp = y + (size_t)row * ldy + offy + g * 4;
    float4 o;
    if (mul) {
        const float4 m4 = *(const float4*)(mul + (size_t)row * 32 + g * 4);
        o.x = a * inv * m4.x; o.y = b * inv * m4.y;
        o.z = c * inv * m4.z; o.w = d * inv * m4.w;
    } else {
        o.x = a * inv; o.y = b * inv; o.z = c * inv; o.w = d * inv;
    }
    *(float4*)yp = o;
}

__global__ void k_ln_gelu(float* __restrict__ x, const float* __restrict__ g,
                          const float* __restrict__ b, int nrow) {
    int row = blockIdx.x * blockDim.x + threadIdx.x;
    if (row >= nrow) return;
    float* xr = x + (size_t)row * 32;
    float m = 0.0f;
#pragma unroll
    for (int c = 0; c < 32; c++) m += xr[c];
    m *= (1.0f / 32.0f);
    float v = 0.0f;
#pragma unroll
    for (int c = 0; c < 32; c++) { float d = xr[c] - m; v += d * d; }
    v *= (1.0f / 32.0f);
    float inv = rsqrtf(v + 1e-5f);
#pragma unroll
    for (int c = 0; c < 32; c++) xr[c] = geluf((xr[c] - m) * inv * g[c] + b[c]);
}

// tar/con construction + 64ch concat [con|tar] for ker linear
__global__ void k_build_tarcon(const float* __restrict__ Z, int ldz, int offz,
                               const float* __restrict__ pos, int s,
                               float* __restrict__ tar, float* __restrict__ con,
                               float* __restrict__ ct, int nrow) {
    int row = blockIdx.x * blockDim.x + threadIdx.x;
    if (row >= nrow) return;
    int l = row % LLEN, t = (row / LLEN) & 3;
    bool mask = (l >= s) && (l < s + 6);
#pragma unroll
    for (int c = 0; c < 32; c++) {
        float zv = Z[(size_t)row * ldz + offz + c];
        float pv = pos[(c * 4 + t) * LLEN + l];
        float tv = mask ? zv : pv;
        float cv = mask ? pv : zv;
        tar[(size_t)row * 32 + c] = tv;
        con[(size_t)row * 32 + c] = cv;
        ct[(size_t)row * 64 + c] = cv;
        ct[(size_t)row * 64 + 32 + c] = tv;
    }
}

// [con2 | r] concat for p_conv  (con2 = mask ? pre : con)
__global__ void k_build_cat2(const float* __restrict__ con, const float* __restrict__ r,
                             const float* __restrict__ pre, int s,
                             float* __restrict__ ct, int nrow) {
    int row = blockIdx.x * blockDim.x + threadIdx.x;
    if (row >= nrow) return;
    int l = row % LLEN, t = (row / LLEN) & 3;
    bool mask = (l >= s) && (l < s + 6);
#pragma unroll
    for (int c = 0; c < 32; c++) {
        float cv = con[(size_t)row * 32 + c];
        float pv = pre[(c * 4 + t) * LLEN + l];
        ct[(size_t)row * 64 + c] = mask ? pv : cv;
        ct[(size_t)row * 64 + 32 + c] = r[(size_t)row * 32 + c];
    }
}

// rule (1,32,4,24) -> channels-last 96x32 rows (t*24+l)
__global__ void k_rul96(const float* __restrict__ rule, float* __restrict__ y) {
    int tid = blockIdx.x * blockDim.x + threadIdx.x;
    if (tid >= 96 * 32) return;
    int c = tid & 31, row = tid >> 5;
    int t = row / LLEN, l = row % LLEN;
    y[tid] = rule[(c * 4 + t) * LLEN + l];
}

// softmax attention over L=24, D=4 per (b,t,head,row); v optionally b-broadcast
__global__ void k_attn(const float* __restrict__ q, const float* __restrict__ k,
                       const float* __restrict__ v, float* __restrict__ out,
                       int total, int vBroad) {
    int tid = blockIdx.x * blockDim.x + threadIdx.x;
    if (tid >= total) return;
    int i = tid % LLEN;
    int hh = (tid / LLEN) % NHEAD;
    int bt = tid / (LLEN * NHEAD);
    const float4 qv = *(const float4*)(q + ((size_t)(bt * LLEN + i)) * 32 + hh * 4);
    float sc[LLEN];
    float mx = -1e30f;
#pragma unroll
    for (int j = 0; j < LLEN; j++) {
        const float4 kv = *(const float4*)(k + ((size_t)(bt * LLEN + j)) * 32 + hh * 4);
        float s = (qv.x * kv.x + qv.y * kv.y + qv.z * kv.z + qv.w * kv.w) * 0.5f;
        sc[j] = s; mx = fmaxf(mx, s);
    }
    float sum = 0.0f;
#pragma unroll
    for (int j = 0; j < LLEN; j++) { sc[j] = expf(sc[j] - mx); sum += sc[j]; }
    float inv = 1.0f / sum;
    float o0 = 0, o1 = 0, o2 = 0, o3 = 0;
    int vbase = vBroad ? ((bt & 3) * LLEN) : (bt * LLEN);
#pragma unroll
    for (int j = 0; j < LLEN; j++) {
        const float4 vv = *(const float4*)(v + ((size_t)(vbase + j)) * 32 + hh * 4);
        float p = sc[j] * inv;
        o0 += p * vv.x; o1 += p * vv.y; o2 += p * vv.z; o3 += p * vv.w;
    }
    float4 o = {o0, o1, o2, o3};
    *(float4*)(out + ((size_t)(bt * LLEN + i)) * 32 + hh * 4) = o;
}

// pl linear over L (24->6) scattered into masked region, passthrough elsewhere
__global__ void k_pl_scatter(const float* __restrict__ pp,
                             const float* __restrict__ Z, int ldz, int offz,
                             const float* __restrict__ plW, const float* __restrict__ plb,
                             int s, float* __restrict__ zout, int total) {
    int tid = blockIdx.x * blockDim.x + threadIdx.x;
    if (tid >= total) return;
    int c = tid & 31, row = tid >> 5;
    int l = row % LLEN, bt = row / LLEN;
    float val;
    if (l >= s && l < s + 6) {
        int j = l - s;
        float acc = plb[j];
        const float* wr = plW + j * LLEN;
#pragma unroll
        for (int lp = 0; lp < LLEN; lp++)
            acc += pp[((size_t)(bt * LLEN + lp)) * 32 + c] * wr[lp];
        val = acc;
    } else {
        val = Z[(size_t)row * ldz + offz + c];
    }
    zout[(size_t)row * 32 + c] = val;
}

// (B,T,L,C)[gz cols 0..31] -> (B,L,C,T) for the T-as-channels convs
__global__ void k_tr_fwd(const float* __restrict__ gz, float* __restrict__ gT, int total) {
    int tid = blockIdx.x * blockDim.x + threadIdx.x;
    if (tid >= total) return;
    int c = tid & 31, row = tid >> 5;
    int l = row % LLEN, bt = row / LLEN;
    int t = bt & 3, b = bt >> 2;
    gT[((size_t)(b * LLEN + l) * 32 + c) * 4 + t] = gz[(size_t)row * 64 + c];
}

__global__ void k_tr_back(const float* __restrict__ gc, float* __restrict__ g2, int total) {
    int tid = blockIdx.x * blockDim.x + threadIdx.x;
    if (tid >= total) return;
    int c = tid & 31, row = tid >> 5;
    int l = row % LLEN, bt = row / LLEN;
    int t = bt & 3, b = bt >> 2;
    g2[tid] = gc[((size_t)(b * LLEN + l) * 32 + c) * 4 + t];
}

// out[b,c,t,l] (NCHW fp32) = relu(x0 - xp), channels-last inputs
__global__ void k_final(const float* __restrict__ x0, const float* __restrict__ xp,
                        float* __restrict__ out, int total) {
    int tid = blockIdx.x * blockDim.x + threadIdx.x;
    if (tid >= total) return;
    int c = tid & 31, row = tid >> 5;
    int l = row % LLEN, bt = row / LLEN;
    int t = bt & 3, b = bt >> 2;
    out[((size_t)(b * 32 + c) * 4 + t) * LLEN + l] = fmaxf(x0[tid] - xp[tid], 0.0f);
}

// ---------------------------------------------------------------------------
// Host orchestration
// ---------------------------------------------------------------------------
static inline int ceildiv(int a, int b) { return (a + b - 1) / b; }

static void gemm(hipStream_t st, const float* X, int ldx, const float* Wt,
                 const float* bias, const float* gain, const float* add, int ldadd,
                 float* Y, int ldy, int colOff, int Nrow, int I, int O, int flags) {
    int waves = (Nrow / 16) * (O / 16);
    k_gemm<<<ceildiv(waves * 32, 256), 256, 0, st>>>(X, ldx, Wt, bias, gain, add,
                                                     ldadd, Y, ldy, colOff, Nrow, I, O, flags);
}

static void conv3(hipStream_t st, const float* X, const float* Wt, const float* gain,
                  const float* bias, const float* add, float* Y, _Float16* Wp,
                  int NB, int H, int Wd, int Cin, int Cout, int flags) {
    const int K = 9 * Cin;
    const int Kp = (K + 31) & ~31;
    const int CoutP = (Cout + 15) & ~15;
    const int tot = CoutP * Kp;
    k_packw<<<ceildiv(tot, 256), 256, 0, st>>>(Wt, Wp, Cin, Cout, Kp, tot);
    int waves = ((NB * H * Wd) / 16) * (CoutP / 16);
    int grid = ceildiv(waves * 32, 256);
    if ((Cin & 31) == 0) {
        k_conv3_c32<<<grid, 256, 0, st>>>(X, Wp, gain, bias, add, Y,
                                          NB, H, Wd, Cin, Cout, Kp, flags);
    } else {
        int lg = (Cin == 4) ? 2 : 4;
        k_conv3_small<<<grid, 256, 0, st>>>(X, Wp, gain, bias, add, Y,
                                            NB, H, Wd, Cin, Cout, Kp, lg, flags);
    }
}

extern "C" void kernel_launch(void* const* d_in, const int* in_sizes, int n_in,
                              void* d_out, int out_size, void* d_ws, size_t ws_size,
                              hipStream_t stream) {
    auto F = [&](int i) { return (const float*)d_in[i]; };
    (void)in_sizes; (void)n_in; (void)out_size; (void)ws_size;

    const int R = BB * TT * LLEN;        // 49152 positions
    const int R32 = R * 32;

    // pytree leaf indices (sorted-key DFS after d_in[0]=x)
    const int ATT0 = 1;                  // 6 leaves each: kv_W,kv_b,m_W,m_b,q_W,q_b
    const int FUSE_W = 25, FUSE_B = 26, FUSE_G = 27;
    const int GP0 = 28, GPN = 41;
    const int MW = 684, MB = 685;
    const int flagStart[5] = {0, 18, 12, 6, 0};
    const int chains[4][4] = {{1, 2, 3, 4}, {2, 3, 1, 4}, {3, 2, 4, 1}, {4, 3, 2, 1}};

    // workspace layout (floats)
    float* w = (float*)d_ws;
    float* x0 = w;  w += R32;
    float* xi = w;  w += R32;
    float* cat = w; w += (size_t)R * 128;
    float* P[15];
    for (int i = 0; i < 15; i++) { P[i] = w; w += R32; }
    float* gz  = w; w += (size_t)R * 64;
    float* ct  = w; w += (size_t)R * 64;
    float* big = w; w += (size_t)R * 128;    // aliases gc1 (Rg*16) and c2a out
    float* rul96 = w; w += 96 * 32;
    float* v96   = w; w += 96 * 32;
    _Float16* Wp = (_Float16*)w; w += 24 * 1024;   // 48K halves: packed conv weights

    // 1) input projection over LQ: (B,C,T,25) -> channels-last (B,T,L,32)
    k_proj_in<<<ceildiv(R32, 256), 256, 0, stream>>>(F(0), F(MW), F(MB), x0, R32);

    for (int ci = 0; ci < 4; ci++) {
        k_copy<<<ceildiv(R32, 256), 256, 0, stream>>>(x0, xi, R32);

        for (int j = 0; j < 4; j++) {
            const int G = GP0 + GPN * (4 * ci + j);
            const int s = flagStart[chains[ci][j]];

            // sc = cna(xi, down, act=False)  (1x1 conv => GEMM, *g+b)
            gemm(stream, xi, 32, F(G + 12), F(G + 13), F(G + 14), nullptr, 0,
                 P[1], 32, 0, R, 32, 32, EP_SCALE);
            // xn = l2n(xi)
            k_l2n_row<<<ceildiv(R, 256), 256, 0, stream>>>(xi, P[0], R);
            // gz = lp1(xn)  (32 -> 64)
            gemm(stream, P[0], 32, F(G + 17), F(G + 18), nullptr, nullptr, 0,
                 gz, 64, 0, R, 32, 64, EP_BIAS);

            // ---- g path: convs over (L=24, C=32) grid with T as channels ----
            k_tr_fwd<<<ceildiv(R32, 256), 256, 0, stream>>>(gz, P[2], R32);
            conv3(stream, P[2], F(G + 0), F(G + 2), F(G + 1), nullptr, big, Wp,
                  BB, LLEN, 32, 4, 16, EP_RELU);                         // c1a
            conv3(stream, big, F(G + 3), F(G + 5), F(G + 4), nullptr, P[3], Wp,
                  BB, LLEN, 32, 16, 4, EP_RELU);                         // c1b
            k_tr_back<<<ceildiv(R32, 256), 256, 0, stream>>>(P[3], P[4], R32);
            gemm(stream, P[4], 32, F(G + 15), F(G + 16), nullptr, nullptr, 0,
                 P[5], 32, 0, R, 32, 32, EP_BIAS);                       // gm

            // ---- z path: pred_attn on z = gz[:,32:64] ----
            k_build_tarcon<<<ceildiv(R, 256), 256, 0, stream>>>(
                gz, 64, 32, F(G + 34), s, P[6], P[7], ct, R);
            gemm(stream, ct, 64, F(G + 23), F(G + 24), nullptr, nullptr, 0,
                 P[8], 32, 0, R, 64, 32, EP_BIAS);                       // ker
            k_ln_gelu<<<ceildiv(R, 256), 256, 0, stream>>>(P[8], F(G + 26), F(G + 25), R);
            k_l2n_head<<<ceildiv(R * 8, 256), 256, 0, stream>>>(
                P[8], 32, 0, nullptr, P[8], 32, 0, R);                   // tc
            gemm(stream, P[6], 32, F(G + 36), F(G + 37), nullptr, nullptr, 0,
                 P[9], 32, 0, R, 32, 32, EP_BIAS);                       // q lin
            k_l2n_head<<<ceildiv(R * 8, 256), 256, 0, stream>>>(
                P[9], 32, 0, P[8], P[9], 32, 0, R);                      // q = l2n*tc
            gemm(stream, P[7], 32, F(G + 21), F(G + 22), nullptr, nullptr, 0,
                 P[10], 32, 0, R, 32, 32, EP_BIAS);                      // k lin
            k_l2n_head<<<ceildiv(R * 8, 256), 256, 0, stream>>>(
                P[10], 32, 0, P[8], P[10], 32, 0, R);                    // k = l2n*tc
            k_rul96<<<ceildiv(96 * 32, 256), 256, 0, stream>>>(F(G + 38), rul96);
            gemm(stream, rul96, 32, F(G + 39), F(G + 40), nullptr, nullptr, 0,
                 v96, 32, 0, 96, 32, 32, EP_BIAS);                       // v (b-invariant)
            k_l2n_head<<<ceildiv(96 * 8, 256), 256, 0, stream>>>(
                v96, 32, 0, nullptr, v96, 32, 0, 96);
            k_attn<<<ceildiv(BB * TT * NHEAD * LLEN, 256), 256, 0, stream>>>(
                P[9], P[10], v96, P[11], BB * TT * NHEAD * LLEN, 1);
            gemm(stream, P[11], 32, F(G + 27), F(G + 28), nullptr, nullptr, 0,
                 P[12], 32, 0, R, 32, 32, EP_BIAS);                      // r = m(attn)
            k_build_cat2<<<ceildiv(R, 256), 256, 0, stream>>>(
                P[7], P[12], F(G + 35), s, ct, R);
            conv3(stream, ct, F(G + 29), F(G + 31), F(G + 30), nullptr, P[13], Wp,
                  BB, TT, LLEN, 64, 32, EP_RELU);                        // p_conv
            k_pl_scatter<<<ceildiv(R32, 256), 256, 0, stream>>>(
                P[13], gz, 64, 32, F(G + 32), F(G + 33), s, P[14], R32); // zout

            // ---- merge: o = lp2(gelu(gm)*z); c2a; c2b; + sc ----
            k_gelumul<<<ceildiv(R32, 256), 256, 0, stream>>>(P[5], P[14], P[4], R32);
            gemm(stream, P[4], 32, F(G + 19), F(G + 20), nullptr, nullptr, 0,
                 P[6], 32, 0, R, 32, 32, EP_BIAS);                       // lp2
            conv3(stream, P[6], F(G + 6), F(G + 8), F(G + 7), nullptr, big, Wp,
                  BB, TT, LLEN, 32, 128, EP_RELU);                       // c2a
            conv3(stream, big, F(G + 9), F(G + 11), F(G + 10), P[1], xi, Wp,
                  BB, TT, LLEN, 128, 32, EP_RELU);                       // c2b + sc
        }

        // ---- cro_attn(e = relu(x0 - xi), xi); write slice of cat ----
        const int A = ATT0 + 6 * ci;
        k_relu_sub<<<ceildiv(R32, 256), 256, 0, stream>>>(x0, xi, P[0], R32);
        gemm(stream, P[0], 32, F(A + 4), F(A + 5), nullptr, nullptr, 0,
             P[9], 32, 0, R, 32, 32, EP_BIAS);                           // q
        k_l2n_head<<<ceildiv(R * 8, 256), 256, 0, stream>>>(
            P[9], 32, 0, nullptr, P[9], 32, 0, R);
        gemm(stream, xi, 32, F(A + 0), F(A + 1), nullptr, nullptr, 0,
             ct, 64, 0, R, 32, 64, EP_BIAS);                             // kv
        k_l2n_head<<<ceildiv(R * 8, 256), 256, 0, stream>>>(
            ct, 64, 0, nullptr, P[10], 32, 0, R);                        // k
        k_l2n_head<<<ceildiv(R * 8, 256), 256, 0, stream>>>(
            ct, 64, 32, nullptr, P[8], 32, 0, R);                        // v
        k_attn<<<ceildiv(BB * TT * NHEAD * LLEN, 256), 256, 0, stream>>>(
            P[9], P[10], P[8], P[11], BB * TT * NHEAD * LLEN, 0);
        gemm(stream, P[11], 32, F(A + 2), F(A + 3), nullptr, xi, 32,
             cat, 128, 32 * ci, R, 32, 32, EP_BIAS | EP_ADD);            // m + residual
    }

    // fuse: 1x1 conv 128->32 with *g+b, relu;  out = relu(x0 - xp) as NCHW
    gemm(stream, cat, 128, F(FUSE_W), F(FUSE_B), F(FUSE_G), nullptr, 0,
         P[0], 32, 0, R, 128, 32, EP_SCALE | EP_RELU);
    k_final<<<ceildiv(R32, 256), 256, 0, stream>>>(x0, P[0], (float*)d_out, R32);
}
